// DCN_off_b_50070728737063
// MI455X (gfx1250) — matrix operand order repaired and checked by reading, not compiled, not run on hardware
//
#include <hip/hip_runtime.h>
#include <hip/hip_bf16.h>

// ---------------------------------------------------------------------------
// Problem constants (from reference): Fc=64, dg=8, B=4, H=W=160.
// ---------------------------------------------------------------------------
#define BATCH 4
#define HDIM  160
#define WDIM  160
#define HW    (HDIM * WDIM)      // 25600
#define FC    64
#define DG    8

typedef __attribute__((ext_vector_type(16))) __bf16 v16bf;
typedef __attribute__((ext_vector_type(8)))  float  v8f;

union FragU {
    uint4 q[2];
    v16bf v;
};

// round-to-nearest-even fp32 -> bf16 (bit pattern in ushort)
__device__ __forceinline__ unsigned short f2bf(float f) {
    union { float f; unsigned u; } x;
    x.f = f;
    unsigned u = x.u;
    unsigned r = (u + 0x7FFFu + ((u >> 16) & 1u)) >> 16;
    return (unsigned short)r;
}

// ---------------------------------------------------------------------------
// Weight pack: fp32 [O][I][3][3] -> bf16 CDNA5 A-fragment stream
// layout: [oc][kc][tap][lane=32][j=16] halves, O padded to Opad.
// A-matrix (16x32 bf16) lane mapping used by this code (contiguous k, B matches): M = lane%16, kh = lane/16,
//   element j -> K = 16*kh + j
// ---------------------------------------------------------------------------
__global__ void pack_wmma_weights(const float* __restrict__ w,
                                  unsigned short* __restrict__ dst,
                                  int Oreal, int Opad, int I, int total) {
    int e = blockIdx.x * blockDim.x + threadIdx.x;
    if (e >= total) return;
    int j   = e & 15;
    int L   = (e >> 4) & 31;
    int t   = (e >> 9) % 9;
    int rst = e / 4608;          // 512*9
    int nkc = I >> 5;
    int kc  = rst % nkc;
    int oc  = rst / nkc;
    int m   = L & 15;
    int kh  = L >> 4;
    int k   = 16 * kh + j;
    int o   = oc * 16 + m;
    int i   = kc * 32 + k;
    int ky  = t / 3, kx = t % 3;
    float val = 0.f;
    if (o < Oreal) val = w[((size_t)(o * I + i) * 3 + ky) * 3 + kx];
    dst[e] = f2bf(val);
}

// DCN weight repack: [O=64][C=64][9] -> [c][kk][o] (contiguous over o)
__global__ void pack_dcn_weights(const float* __restrict__ w,
                                 float* __restrict__ dst) {
    int e = blockIdx.x * blockDim.x + threadIdx.x;
    if (e >= 64 * 9 * 64) return;
    int o  = e & 63;
    int kk = (e >> 6) % 9;
    int c  = e / 576;
    dst[e] = w[((size_t)(o * 64 + c)) * 9 + kk];
}

// ---------------------------------------------------------------------------
// Implicit-GEMM 3x3 conv (stride 1, SAME pad) via v_wmma_f32_16x16x32_bf16.
// Workgroup: one 16-pixel row tile (b, h, w0..w0+15), (Opad/16) waves,
// each wave owns a 16-output-channel tile.
// LDS halo tile: [dy(3)][w(18, pad 20)][ch(32, pad 40)] bf16 halves.
// IN_MODE: 0 = concat(ali,ref) fp32; 1 = bf16 tensor.
// OUT_MODE: 0 = fp32; 1 = fp32 + bf16; 2 = bf16 only.
// ACT: 0 = none; 1 = leaky-relu(0.1); 2 = sigmoid for channels >= 144.
// ---------------------------------------------------------------------------
template <int IN_MODE, int OUT_MODE, int ACT>
__global__ void conv3x3_wmma(const float* __restrict__ inA,
                             const float* __restrict__ inB,
                             const unsigned short* __restrict__ inH,
                             const unsigned short* __restrict__ wpack,
                             const float* __restrict__ bias,
                             float* __restrict__ outF,
                             unsigned short* __restrict__ outH,
                             int Cin, int Oreal) {
    __shared__ unsigned short smem[3 * 20 * 40];

    const int tile = blockIdx.x;
    const int w0   = (tile % (WDIM / 16)) * 16;
    int tmp        = tile / (WDIM / 16);
    const int h    = tmp % HDIM;
    const int b    = tmp / HDIM;

    const int lane = threadIdx.x & 31;
    const int wave = threadIdx.x >> 5;
    const int kh   = lane >> 4;   // K-half selector
    const int n    = lane & 15;   // pixel column within tile
    const int nkc  = Cin >> 5;    // 32-channel K chunks

    v8f acc = {0.f, 0.f, 0.f, 0.f, 0.f, 0.f, 0.f, 0.f};

    for (int kc = 0; kc < nkc; ++kc) {
        __syncthreads();
        // ---- cooperative halo stage: 32 channels x 3 rows x 18 cols ----
        for (int idx = threadIdx.x; idx < 32 * 54; idx += blockDim.x) {
            int il  = idx / 54;
            int rem = idx % 54;
            int dy  = rem / 18;
            int wi  = rem % 18;
            int hh  = h - 1 + dy;
            int ww  = w0 - 1 + wi;
            unsigned short hv = 0;
            if (hh >= 0 && hh < HDIM && ww >= 0 && ww < WDIM) {
                int ch = kc * 32 + il;
                if constexpr (IN_MODE == 0) {
                    float f = (ch < FC)
                        ? inA[((size_t)(b * FC + ch) * HDIM + hh) * WDIM + ww]
                        : inB[((size_t)(b * FC + (ch - FC)) * HDIM + hh) * WDIM + ww];
                    hv = f2bf(f);
                } else {
                    hv = inH[((size_t)(b * Cin + ch) * HDIM + hh) * WDIM + ww];
                }
            }
            smem[(dy * 20 + wi) * 40 + il] = hv;
        }
        __syncthreads();

        // ---- 9 taps of 16x16x32 bf16 WMMA ----
        const unsigned short* wbase =
            wpack + ((size_t)(wave * nkc + kc) * 9) * 512 + lane * 16;
        if (kc + 1 < nkc)
            __builtin_prefetch(wpack + ((size_t)(wave * nkc + kc + 1) * 9) * 512, 0, 3);
#pragma unroll
        for (int t = 0; t < 9; ++t) {
            FragU a, bf;
            const uint4* ag = reinterpret_cast<const uint4*>(wbase + t * 512);
            a.q[0] = ag[0];
            a.q[1] = ag[1];
            const int dy = t / 3;
            const int kx = t % 3;
            const uint4* bs =
                reinterpret_cast<const uint4*>(&smem[(dy * 20 + n + kx) * 40 + kh * 16]);
            bf.q[0] = bs[0];
            bf.q[1] = bs[1];
            acc = __builtin_amdgcn_wmma_f32_16x16x32_bf16(
                false, a.v, false, bf.v, (short)0, acc, false, false);
        }
    }

    // ---- epilogue: D[m=o, n=pixel]; lane->pixel gives coalesced stores ----
#pragma unroll
    for (int r = 0; r < 8; ++r) {
        int o = wave * 16 + r + 8 * kh;
        if (o < Oreal) {
            float v = acc[r] + bias[o];
            if constexpr (ACT == 1) v = (v >= 0.f) ? v : 0.1f * v;
            if constexpr (ACT == 2) {
                if (o >= 144) v = 1.f / (1.f + __expf(-v));
            }
            size_t gi = ((size_t)(b * Oreal + o) * HDIM + h) * WDIM + w0 + n;
            if constexpr (OUT_MODE == 0 || OUT_MODE == 1) outF[gi] = v;
            if constexpr (OUT_MODE == 1 || OUT_MODE == 2) outH[gi] = f2bf(v);
        }
    }
}

// ---------------------------------------------------------------------------
// Modulated deformable conv v2 (fp32 VALU): 1 thread per output pixel,
// 64 fp32 accumulators, bilinear sample per (group, tap, channel),
// broadcast float4 weight loads, fused bias + leaky-relu.
// com layout: [B][216][H][W] with ch 0..71 = dy, 72..143 = dx,
// 144..215 = sigmoid(mask) (already applied by conv4 epilogue).
// ---------------------------------------------------------------------------
__global__ void dcn_mod_deform(const float* __restrict__ xs,
                               const float* __restrict__ com,
                               const float* __restrict__ wp,   // [c][kk][64]
                               const float* __restrict__ bias,
                               float* __restrict__ out) {
    int p = blockIdx.x * blockDim.x + threadIdx.x;
    if (p >= BATCH * HW) return;
    int b   = p / HW;
    int rem = p % HW;
    int h   = rem / WDIM;
    int w   = rem % WDIM;

    float acc[64];
#pragma unroll
    for (int o = 0; o < 64; ++o) acc[o] = 0.f;

    const size_t comb = (size_t)b * 216 * HW + rem;

    for (int g = 0; g < DG; ++g) {
        for (int kk = 0; kk < 9; ++kk) {
            const int ky = kk / 3 - 1;
            const int kx = kk % 3 - 1;
            const int chy = g * 9 + kk;
            float offy = com[comb + (size_t)chy * HW];
            float offx = com[comb + (size_t)(72 + chy) * HW];
            float mval = com[comb + (size_t)(144 + chy) * HW];

            float py = offy + (float)(h + ky);
            float px = offx + (float)(w + kx);
            float y0f = floorf(py);
            float x0f = floorf(px);
            float wy = py - y0f;
            float wx = px - x0f;
            int y0 = (int)y0f, x0 = (int)x0f;
            int y1 = y0 + 1,   x1 = x0 + 1;

            float vy0 = (y0 >= 0 && y0 < HDIM) ? 1.f : 0.f;
            float vy1 = (y1 >= 0 && y1 < HDIM) ? 1.f : 0.f;
            float vx0 = (x0 >= 0 && x0 < WDIM) ? 1.f : 0.f;
            float vx1 = (x1 >= 0 && x1 < WDIM) ? 1.f : 0.f;
            int cy0 = min(max(y0, 0), HDIM - 1);
            int cy1 = min(max(y1, 0), HDIM - 1);
            int cx0 = min(max(x0, 0), WDIM - 1);
            int cx1 = min(max(x1, 0), WDIM - 1);

            float w00 = (1.f - wy) * (1.f - wx) * vy0 * vx0 * mval;
            float w01 = (1.f - wy) * wx         * vy0 * vx1 * mval;
            float w10 = wy         * (1.f - wx) * vy1 * vx0 * mval;
            float w11 = wy         * wx         * vy1 * vx1 * mval;

            int i00 = cy0 * WDIM + cx0;
            int i01 = cy0 * WDIM + cx1;
            int i10 = cy1 * WDIM + cx0;
            int i11 = cy1 * WDIM + cx1;

            for (int c = 0; c < 8; ++c) {
                const float* xb = xs + (size_t)(b * FC + g * 8 + c) * HW;
                float v = w00 * xb[i00] + w01 * xb[i01] +
                          w10 * xb[i10] + w11 * xb[i11];
                const float4* wrow =
                    reinterpret_cast<const float4*>(wp + ((size_t)((g * 8 + c) * 9 + kk) << 6));
#pragma unroll
                for (int o4 = 0; o4 < 16; ++o4) {
                    float4 wv = wrow[o4];
                    acc[o4 * 4 + 0] = fmaf(v, wv.x, acc[o4 * 4 + 0]);
                    acc[o4 * 4 + 1] = fmaf(v, wv.y, acc[o4 * 4 + 1]);
                    acc[o4 * 4 + 2] = fmaf(v, wv.z, acc[o4 * 4 + 2]);
                    acc[o4 * 4 + 3] = fmaf(v, wv.w, acc[o4 * 4 + 3]);
                }
            }
        }
    }

#pragma unroll
    for (int o = 0; o < 64; ++o) {
        float v = acc[o] + bias[o];
        v = (v >= 0.f) ? v : 0.1f * v;
        out[(size_t)(b * 64 + o) * HW + rem] = v;
    }
}

// ---------------------------------------------------------------------------
// Host launcher
// ---------------------------------------------------------------------------
extern "C" void kernel_launch(void* const* d_in, const int* in_sizes, int n_in,
                              void* d_out, int out_size, void* d_ws, size_t ws_size,
                              hipStream_t stream) {
    (void)in_sizes; (void)n_in; (void)out_size; (void)ws_size;

    const float* ali    = (const float*)d_in[0];
    const float* ref    = (const float*)d_in[1];
    const float* w_conv = (const float*)d_in[2];
    const float* b_conv = (const float*)d_in[3];
    const float* w_off  = (const float*)d_in[4];
    const float* b_off  = (const float*)d_in[5];
    const float* w_x    = (const float*)d_in[6];
    const float* b_x    = (const float*)d_in[7];
    const float* w_com  = (const float*)d_in[8];
    const float* b_com  = (const float*)d_in[9];
    const float* w_dcn  = (const float*)d_in[10];
    const float* b_dcn  = (const float*)d_in[11];

    float* out_main = (float*)d_out;                              // [B,64,H,W]
    float* out_offf = (float*)d_out + (size_t)BATCH * FC * HW;    // [B,64,H,W]

    // ---- carve workspace ----
    char* ws = (char*)d_ws;
    size_t off = 0;
    auto carve = [&](size_t bytes) {
        char* p = ws + off;
        off = (off + bytes + 255) & ~(size_t)255;
        return p;
    };
    unsigned short* tensor_h  = (unsigned short*)carve((size_t)BATCH * 128 * HW * 2);
    unsigned short* offfeat_h = (unsigned short*)carve((size_t)BATCH * 64 * HW * 2);
    float*          x_f       = (float*)carve((size_t)BATCH * 64 * HW * 4);
    float*          com_f     = (float*)carve((size_t)BATCH * 216 * HW * 4);
    unsigned short* pk1       = (unsigned short*)carve((size_t)128 * 128 * 9 * 2);
    unsigned short* pk2       = (unsigned short*)carve((size_t)64 * 128 * 9 * 2);
    unsigned short* pk3       = (unsigned short*)carve((size_t)64 * 128 * 9 * 2);
    unsigned short* pk4       = (unsigned short*)carve((size_t)224 * 64 * 9 * 2);
    float*          dcnw      = (float*)carve((size_t)64 * 9 * 64 * 4);

    // ---- weight packing ----
    {
        int t1 = 128 * 128 * 9;
        pack_wmma_weights<<<(t1 + 255) / 256, 256, 0, stream>>>(w_conv, pk1, 128, 128, 128, t1);
        int t2 = 64 * 128 * 9;
        pack_wmma_weights<<<(t2 + 255) / 256, 256, 0, stream>>>(w_off, pk2, 64, 64, 128, t2);
        pack_wmma_weights<<<(t2 + 255) / 256, 256, 0, stream>>>(w_x, pk3, 64, 64, 128, t2);
        int t4 = 224 * 64 * 9;
        pack_wmma_weights<<<(t4 + 255) / 256, 256, 0, stream>>>(w_com, pk4, 216, 224, 64, t4);
        int t5 = 64 * 9 * 64;
        pack_dcn_weights<<<(t5 + 255) / 256, 256, 0, stream>>>(w_dcn, dcnw);
    }

    const int tiles = BATCH * HDIM * (WDIM / 16);   // 6400 pixel tiles

    // conv1: concat(ali,ref) [128] -> tensor [128], lrelu, bf16 out
    conv3x3_wmma<0, 2, 1><<<tiles, 256, 0, stream>>>(
        ali, ref, nullptr, pk1, b_conv, nullptr, tensor_h, 128, 128);

    // conv2: tensor [128] -> offset_feat [64], lrelu, fp32(d_out)+bf16(ws)
    conv3x3_wmma<1, 1, 1><<<tiles, 128, 0, stream>>>(
        nullptr, nullptr, tensor_h, pk2, b_off, out_offf, offfeat_h, 128, 64);

    // conv3: tensor [128] -> x [64], lrelu, fp32
    conv3x3_wmma<1, 0, 1><<<tiles, 128, 0, stream>>>(
        nullptr, nullptr, tensor_h, pk3, b_x, x_f, nullptr, 128, 64);

    // conv4: offset_feat [64] -> offsets+mask [216], sigmoid on ch>=144, fp32
    conv3x3_wmma<1, 0, 2><<<tiles, 448, 0, stream>>>(
        nullptr, nullptr, offfeat_h, pk4, b_com, com_f, nullptr, 64, 216);

    // DCN: modulated deformable conv + bias + lrelu -> d_out (first half)
    dcn_mod_deform<<<(BATCH * HW + 255) / 256, 256, 0, stream>>>(
        x_f, com_f, dcnw, b_dcn, out_main);
}
